// MoELayer_26096221290607
// MI455X (gfx1250) — matrix operand order repaired and checked by reading, not compiled, not run on hardware
//
#include <hip/hip_runtime.h>
#include <hip/hip_bf16.h>

// ---------------- problem constants (B=4,S=2048,H=1024,I=2816,E=8) ----------
#define TOKS 8192          // B*S
#define HD   1024
#define ID   2816
#define NE   8
#define SEQ  2048
#define NB   4

// GEMM tiling: 128x128 block tile, 8 wave32 waves, each wave 64x32 (4x2 wmma)
#define LDA  40            // padded LDS stride (bf16 elems) for A   (32 + 8)
#define LDB  40            // padded LDS stride (bf16 elems) for B^T (32 + 8)

// CDNA5 async global->LDS path (ASYNCcnt), guarded so either toolchain compiles
#if defined(__has_builtin)
#  if __has_builtin(__builtin_amdgcn_global_load_async_to_lds_b128)
#    define HAVE_ASYNC_LDS 1
#  endif
#  if __has_builtin(__builtin_amdgcn_s_wait_asynccnt)
#    define HAVE_WAIT_ASYNC 1
#  endif
#endif

typedef __attribute__((ext_vector_type(16))) __bf16 v16bf;
typedef __attribute__((ext_vector_type(8)))  __bf16 bf16x8;
typedef __attribute__((ext_vector_type(4)))  __bf16 bf16x4;
typedef __attribute__((ext_vector_type(8)))  float  v8f;
typedef __attribute__((ext_vector_type(4)))  float  f32x4;

// async-to-LDS builtin wants pointer-to-16B-vector operands (per diagnostic):
//   param0: vector_size(16) int, addrspace(1)   param1: same pointee, LDS
typedef int v4i_t __attribute__((vector_size(16)));
typedef __attribute__((address_space(1))) v4i_t* gptr_b128;
typedef __attribute__((address_space(3))) v4i_t* lptr_b128;

__device__ __forceinline__ v16bf make16(bf16x8 lo, bf16x8 hi) {
  return __builtin_shufflevector(lo, hi, 0,1,2,3,4,5,6,7,8,9,10,11,12,13,14,15);
}

// A fragment, 16x32 bf16, ISA layout: lane m=l&15, khalf=(l>>4);
// element e -> K = 8*khalf + (e&7) + 16*(e>>3)  => two contiguous 8-elem runs.
__device__ __forceinline__ v16bf load_a_frag(const __bf16* As, int mrow, int lane) {
  const int base = (lane >> 4) * 8;
  const __bf16* p = As + mrow * LDA + base;
  bf16x8 lo = *(const bf16x8*)p;          // K = base .. base+7
  bf16x8 hi = *(const bf16x8*)(p + 16);   // K = base+16 .. base+23
  return make16(lo, hi);
}

// B fragment, 32x16 bf16 from LDS holding B TRANSPOSED as [n][k]:
// lane n=l&15, kbase=8*(l>>4); element e -> K = kbase + (e&7) + 16*(e>>3).
__device__ __forceinline__ v16bf load_b_frag(const __bf16* Bs, int nrow, int lane) {
  const int kb = (lane >> 4) * 8;
  const __bf16* p = Bs + nrow * LDB + kb;
  bf16x8 lo = *(const bf16x8*)p;          // K = kb .. kb+7
  bf16x8 hi = *(const bf16x8*)(p + 16);   // K = kb+16 .. kb+23
  return make16(lo, hi);
}

__device__ __forceinline__ void wait_async() {
#if defined(HAVE_WAIT_ASYNC)
  __builtin_amdgcn_s_wait_asynccnt(0);
#elif defined(HAVE_ASYNC_LDS)
  asm volatile("s_wait_asynccnt 0" ::: "memory");
#endif
}

// ---------------------------------------------------------------------------
// 1) Router: logits = hs @ router_w^T + b, softmax -> routing [TOKS, NE] f32
// one wave per token
// ---------------------------------------------------------------------------
__global__ void __launch_bounds__(256)
router_kernel(const float* __restrict__ hs, const float* __restrict__ rw,
              const float* __restrict__ rb, float* __restrict__ routing) {
  const int lane  = threadIdx.x & 31;
  const int wave  = threadIdx.x >> 5;
  const int token = blockIdx.x * 8 + wave;

  float acc[NE];
#pragma unroll
  for (int e = 0; e < NE; ++e) acc[e] = 0.f;

  const float* hrow = hs + (size_t)token * HD;
  for (int h = lane; h < HD; h += 32) {
    float x = hrow[h];
#pragma unroll
    for (int e = 0; e < NE; ++e) acc[e] += x * rw[e * HD + h];
  }
#pragma unroll
  for (int e = 0; e < NE; ++e) {
#pragma unroll
    for (int off = 16; off > 0; off >>= 1) acc[e] += __shfl_xor(acc[e], off, 32);
    acc[e] += rb[e];
  }
  float mx = acc[0];
#pragma unroll
  for (int e = 1; e < NE; ++e) mx = fmaxf(mx, acc[e]);
  float sum = 0.f, mine = 0.f;
#pragma unroll
  for (int e = 0; e < NE; ++e) {
    float ex = __expf(acc[e] - mx);
    sum += ex;
    if (e == lane) mine = ex;
  }
  if (lane < NE) routing[(size_t)token * NE + lane] = mine / sum;
}

// ---------------------------------------------------------------------------
// 2) balance loss = mean_{s,e}((mean_b routing - 1/E)^2) * 0.01
// ---------------------------------------------------------------------------
__global__ void __launch_bounds__(256)
balance_kernel(const float* __restrict__ routing, float* __restrict__ loss) {
  __shared__ float red[256];
  const int idx = blockIdx.x * 256 + threadIdx.x;   // 0 .. S*E-1
  const int s = idx >> 3, e = idx & 7;
  float m = 0.f;
#pragma unroll
  for (int b = 0; b < NB; ++b) m += routing[((size_t)(b * SEQ + s)) * NE + e];
  m *= (1.f / NB);
  const float d = m - (1.f / NE);
  red[threadIdx.x] = d * d;
  __syncthreads();
  for (int off = 128; off > 0; off >>= 1) {
    if (threadIdx.x < off) red[threadIdx.x] += red[threadIdx.x + off];
    __syncthreads();
  }
  if (threadIdx.x == 0) atomicAdd(loss, red[0] * (0.01f / (float)(SEQ * NE)));
}

// ---------------------------------------------------------------------------
// 3) fused gate/up GEMM per expert:
//    t1[t,n] = silu(hs@gate_w)[t,n] * (hs@up_w)[t,n]   (bf16 out)
//    grid (TOKS/128, ID/128), 256 threads
// ---------------------------------------------------------------------------
__global__ void __launch_bounds__(256)
gateup_kernel(const float* __restrict__ hs, const float* __restrict__ gw,
              const float* __restrict__ uw, __bf16* __restrict__ t1) {
  __shared__ __attribute__((aligned(16))) __bf16 As[128 * LDA];
  __shared__ __attribute__((aligned(16))) __bf16 Bg[128 * LDB];
  __shared__ __attribute__((aligned(16))) __bf16 Bu[128 * LDB];

  const int tid  = threadIdx.x;
  const int lane = tid & 31;
  const int wave = tid >> 5;
  const int wm   = wave >> 2;       // 0..1  (64-row strip)
  const int wn   = wave & 3;        // 0..3  (32-col strip)
  const int m0   = blockIdx.x * 128;
  const int n0   = blockIdx.y * 128;

  v8f accG[4][2], accU[4][2];
  const v8f vzero = {0.f,0.f,0.f,0.f,0.f,0.f,0.f,0.f};
#pragma unroll
  for (int mi = 0; mi < 4; ++mi)
#pragma unroll
    for (int ni = 0; ni < 2; ++ni) { accG[mi][ni] = vzero; accU[mi][ni] = vzero; }

  for (int k0 = 0; k0 < HD; k0 += 32) {
    // ---- stage A tile 128x32 (fp32 -> bf16) ----
    {
      const int col = (tid & 7) * 4;
      int row = tid >> 3;                       // 0..31
#pragma unroll
      for (int it = 0; it < 4; ++it, row += 32) {
        f32x4 v = *(const f32x4*)(hs + (size_t)(m0 + row) * HD + k0 + col);
        bf16x4 b; b[0] = (__bf16)v[0]; b[1] = (__bf16)v[1];
                  b[2] = (__bf16)v[2]; b[3] = (__bf16)v[3];
        *(bf16x4*)&As[row * LDA + col] = b;
      }
    }
    // ---- stage Bg/Bu tiles 32x128, stored transposed [n][k] ----
    {
      const int n = (tid & 31) * 4;
      int k = tid >> 5;                         // 0..7
#pragma unroll
      for (int it = 0; it < 4; ++it, k += 8) {
        f32x4 g = *(const f32x4*)(gw + (size_t)(k0 + k) * ID + n0 + n);
        f32x4 u = *(const f32x4*)(uw + (size_t)(k0 + k) * ID + n0 + n);
#pragma unroll
        for (int j = 0; j < 4; ++j) {
          Bg[(n + j) * LDB + k] = (__bf16)g[j];
          Bu[(n + j) * LDB + k] = (__bf16)u[j];
        }
      }
      if (k0 + 32 < HD) {
        __builtin_prefetch(gw + (size_t)(k0 + 32 + (tid >> 5)) * ID + n0 + n, 0, 1);
        __builtin_prefetch(uw + (size_t)(k0 + 32 + (tid >> 5)) * ID + n0 + n, 0, 1);
      }
    }
    __syncthreads();

    v16bf a[4], bg[2], bu[2];
#pragma unroll
    for (int mi = 0; mi < 4; ++mi)
      a[mi] = load_a_frag(As, wm * 64 + mi * 16 + (lane & 15), lane);
#pragma unroll
    for (int ni = 0; ni < 2; ++ni) {
      bg[ni] = load_b_frag(Bg, wn * 32 + ni * 16 + (lane & 15), lane);
      bu[ni] = load_b_frag(Bu, wn * 32 + ni * 16 + (lane & 15), lane);
    }
#pragma unroll
    for (int mi = 0; mi < 4; ++mi)
#pragma unroll
      for (int ni = 0; ni < 2; ++ni) {
        accG[mi][ni] = __builtin_amdgcn_wmma_f32_16x16x32_bf16(
            false, a[mi], false, bg[ni], (short)0, accG[mi][ni], false, false);
        accU[mi][ni] = __builtin_amdgcn_wmma_f32_16x16x32_bf16(
            false, a[mi], false, bu[ni], (short)0, accU[mi][ni], false, false);
      }
    __syncthreads();
  }

  // epilogue: silu(g)*u -> bf16 t1.  C layout: M = r + 8*(lane>>4), N = lane&15
#pragma unroll
  for (int mi = 0; mi < 4; ++mi)
#pragma unroll
    for (int ni = 0; ni < 2; ++ni) {
      const int col = n0 + wn * 32 + ni * 16 + (lane & 15);
#pragma unroll
      for (int r = 0; r < 8; ++r) {
        const int tok = m0 + wm * 64 + mi * 16 + (lane >> 4) * 8 + r;
        float g = accG[mi][ni][r];
        float u = accU[mi][ni][r];
        float val = (g / (1.f + __expf(-g))) * u;
        t1[(size_t)tok * ID + col] = (__bf16)val;
      }
    }
}

// ---------------------------------------------------------------------------
// 4) down GEMM per expert: out[t,n] += (t1 @ down_w)[t,n] * routing[t,e]
//    grid (TOKS/128, HD/128), 256 threads.
//    Double-buffered LDS; A tile staged with CDNA5 async global->LDS
//    (ASYNCcnt), B tile prefetched into registers then converted.
// ---------------------------------------------------------------------------
__device__ __forceinline__ void stage_a_down(const __bf16* __restrict__ t1,
                                             __bf16* dst, int m0, int k0, int tid) {
  const int col = (tid & 3) * 8;
  int row = tid >> 2;                           // 0..63
#pragma unroll
  for (int it = 0; it < 2; ++it, row += 64) {
    const __bf16* g = t1 + (size_t)(m0 + row) * ID + k0 + col;
    __bf16* l = dst + row * LDA + col;
#if defined(HAVE_ASYNC_LDS)
    __builtin_amdgcn_global_load_async_to_lds_b128(
        (gptr_b128)(void*)g, (lptr_b128)(void*)l, 0, 0);
#else
    *(bf16x8*)l = *(const bf16x8*)g;
#endif
  }
}

__device__ __forceinline__ void load_b_regs_down(const float* __restrict__ dw,
                                                 int n0, int k0, int tid, f32x4 r[4]) {
  const int n = (tid & 31) * 4;
  const int k = tid >> 5;                       // 0..7
#pragma unroll
  for (int it = 0; it < 4; ++it)
    r[it] = *(const f32x4*)(dw + (size_t)(k0 + k + 8 * it) * HD + n0 + n);
}

__device__ __forceinline__ void store_b_lds(__bf16* dst, int tid, const f32x4 r[4]) {
  const int n = (tid & 31) * 4;
  const int k = tid >> 5;
#pragma unroll
  for (int it = 0; it < 4; ++it)
#pragma unroll
    for (int j = 0; j < 4; ++j)
      dst[(n + j) * LDB + (k + 8 * it)] = (__bf16)r[it][j];
}

__global__ void __launch_bounds__(256)
down_kernel(const __bf16* __restrict__ t1, const float* __restrict__ dw,
            const float* __restrict__ routing, int expert,
            float* __restrict__ out) {
  __shared__ __attribute__((aligned(16))) __bf16 As[2][128 * LDA];
  __shared__ __attribute__((aligned(16))) __bf16 Bs[2][128 * LDB];

  const int tid  = threadIdx.x;
  const int lane = tid & 31;
  const int wave = tid >> 5;
  const int wm   = wave >> 2;
  const int wn   = wave & 3;
  const int m0   = blockIdx.x * 128;
  const int n0   = blockIdx.y * 128;

  v8f acc[4][2];
  const v8f vzero = {0.f,0.f,0.f,0.f,0.f,0.f,0.f,0.f};
#pragma unroll
  for (int mi = 0; mi < 4; ++mi)
#pragma unroll
    for (int ni = 0; ni < 2; ++ni) acc[mi][ni] = vzero;

  // prologue: stage tile 0 into buffer 0
  f32x4 breg[4];
  stage_a_down(t1, &As[0][0], m0, 0, tid);
  load_b_regs_down(dw, n0, 0, tid, breg);
  store_b_lds(&Bs[0][0], tid, breg);
  wait_async();
  __syncthreads();

  for (int k0 = 0; k0 < ID; k0 += 32) {
    const int buf  = (k0 >> 5) & 1;
    const int nbuf = buf ^ 1;
    const bool more = (k0 + 32) < ID;

    // issue next tile's memory traffic first (overlaps with WMMA below)
    if (more) {
      stage_a_down(t1, &As[nbuf][0], m0, k0 + 32, tid);
      load_b_regs_down(dw, n0, k0 + 32, tid, breg);
    }

    v16bf a[4], b[2];
#pragma unroll
    for (int mi = 0; mi < 4; ++mi)
      a[mi] = load_a_frag(&As[buf][0], wm * 64 + mi * 16 + (lane & 15), lane);
#pragma unroll
    for (int ni = 0; ni < 2; ++ni)
      b[ni] = load_b_frag(&Bs[buf][0], wn * 32 + ni * 16 + (lane & 15), lane);
#pragma unroll
    for (int mi = 0; mi < 4; ++mi)
#pragma unroll
      for (int ni = 0; ni < 2; ++ni)
        acc[mi][ni] = __builtin_amdgcn_wmma_f32_16x16x32_bf16(
            false, a[mi], false, b[ni], (short)0, acc[mi][ni], false, false);

    if (more) store_b_lds(&Bs[nbuf][0], tid, breg);
    wait_async();
    __syncthreads();
  }

#pragma unroll
  for (int mi = 0; mi < 4; ++mi)
#pragma unroll
    for (int ni = 0; ni < 2; ++ni) {
      const int col = n0 + wn * 32 + ni * 16 + (lane & 15);
#pragma unroll
      for (int r = 0; r < 8; ++r) {
        const int tok = m0 + wm * 64 + mi * 16 + (lane >> 4) * 8 + r;
        const float w = routing[(size_t)tok * NE + expert];
        float* p = out + (size_t)tok * HD + col;
        *p += acc[mi][ni][r] * w;       // experts run sequentially -> no race
      }
    }
}

// ---------------------------------------------------------------------------
extern "C" void kernel_launch(void* const* d_in, const int* in_sizes, int n_in,
                              void* d_out, int out_size, void* d_ws, size_t ws_size,
                              hipStream_t stream) {
  (void)in_sizes; (void)n_in; (void)ws_size;
  const float* hs = (const float*)d_in[0];   // [B,S,H]
  const float* rw = (const float*)d_in[1];   // [E,H]
  const float* rb = (const float*)d_in[2];   // [E]
  const float* gw = (const float*)d_in[3];   // [E,H,I]
  const float* uw = (const float*)d_in[4];   // [E,H,I]
  const float* dw = (const float*)d_in[5];   // [E,I,H]
  float* out = (float*)d_out;                // [B,S,H] + 1 (loss)

  // workspace layout: routing f32 [TOKS,NE] | t1 bf16 [TOKS,ID]
  float*  routing = (float*)d_ws;
  __bf16* t1 = (__bf16*)((char*)d_ws + (size_t)TOKS * NE * sizeof(float));

  (void)hipMemsetAsync(d_out, 0, (size_t)out_size * sizeof(float), stream);

  router_kernel<<<TOKS / 8, 256, 0, stream>>>(hs, rw, rb, routing);
  balance_kernel<<<(SEQ * NE) / 256, 256, 0, stream>>>(
      routing, out + (size_t)out_size - 1);

  for (int e = 0; e < NE; ++e) {
    gateup_kernel<<<dim3(TOKS / 128, ID / 128), 256, 0, stream>>>(
        hs, gw + (size_t)e * HD * ID, uw + (size_t)e * HD * ID, t1);
    down_kernel<<<dim3(TOKS / 128, HD / 128), 256, 0, stream>>>(
        t1, dw + (size_t)e * ID * HD, routing, e, out);
  }
}
